// LanguageModel_81741817578244
// MI455X (gfx1250) — matrix operand order repaired and checked by reading, not compiled, not run on hardware
//
#include <hip/hip_runtime.h>
#include <hip/hip_bf16.h>

// ---------------------------------------------------------------------------
// GRU language model for MI455X (gfx1250, wave32, WMMA).
//   K1: x_gates = embed @ w_ih^T + b_ih     (split-bf16 WMMA, ~fp32 accuracy)
//   K2: 128x sequential GRU steps           (split-bf16 WMMA, ping-pong h)
//   K3: logits = hs @ w_out^T               (bf16 WMMA, double-buffered LDS,
//                                            async global->LDS staging, NT stores)
// Roofline: projection output stores (524 MB @ 23.3 TB/s ~ 22us) are the floor.
// ---------------------------------------------------------------------------

#define B_  32
#define S_  128
#define E_  512
#define H_  512
#define V_  32000
#define G3_ 1536   // 3*H

typedef __attribute__((ext_vector_type(16))) __bf16 v16bf;
typedef __attribute__((ext_vector_type(8)))  float  v8f;

#define LDT 40     // padded LDS row stride in bf16 elems (80B = 20 banks -> conflict free)

union BfPack4 { __bf16 e[4]; uint2 u; };

// gfx1250 async global->LDS loads (ASYNCcnt-tracked), if this clang declares them.
#if __has_builtin(__builtin_amdgcn_global_load_async_to_lds_b128) && \
    __has_builtin(__builtin_amdgcn_s_wait_asynccnt)
#define USE_ASYNC_LDS 1
#else
#define USE_ASYNC_LDS 0
#endif

#if USE_ASYNC_LDS
// Builtin prototype (from clang diagnostic): (int4 AS1*, int4 AS3*, imm int, imm int)
typedef int v4i_ __attribute__((vector_size(16)));
typedef __attribute__((address_space(1))) v4i_* g_v4i_p;
typedef __attribute__((address_space(3))) v4i_* l_v4i_p;
#endif

static __device__ __forceinline__ void copy16_g2l(__bf16* lds_dst, const __bf16* gsrc) {
#if USE_ASYNC_LDS
  __builtin_amdgcn_global_load_async_to_lds_b128((g_v4i_p)gsrc, (l_v4i_p)lds_dst, 0, 0);
#else
  *(uint4*)lds_dst = *(const uint4*)gsrc;
#endif
}

static __device__ __forceinline__ void wait_async_all() {
#if USE_ASYNC_LDS
  __builtin_amdgcn_s_wait_asynccnt(0);
#endif
}

static __device__ __forceinline__ v8f wmma_bf16(v16bf a, v16bf b, v8f c) {
  // modifiers are literal immediates (neg_a, neg_b, c_mod, reuse_a, reuse_b)
  return __builtin_amdgcn_wmma_f32_16x16x32_bf16(false, a, false, b, (short)0, c, false, false);
}

static __device__ __forceinline__ void split4(float4 f, BfPack4& hi, BfPack4& lo) {
  float v[4] = {f.x, f.y, f.z, f.w};
#pragma unroll
  for (int t = 0; t < 4; ++t) {
    __bf16 h = (__bf16)v[t];
    hi.e[t] = h;
    lo.e[t] = (__bf16)(v[t] - (float)h);
  }
}

// A fragment (16x32 bf16): lane L(0..15)=row M, holds K = 16*half + 0..15
// (slots 0..15): two contiguous 16B LDS reads.
static __device__ __forceinline__ v16bf load_a_frag(const __bf16* lds, int mbase, int lane) {
  const int half = lane >> 4, l16 = lane & 15;
  const __bf16* p = lds + (mbase + l16) * LDT + 16 * half;
  v16bf a;
  *(uint4*)&a       = *(const uint4*)p;
  *((uint4*)&a + 1) = *(const uint4*)(p + 8);
  return a;
}

// B fragment (32x16 bf16, B^T row-major [N][K] in LDS): lane L(0..15)=col N,
// holds K = 16*half + 0..15: two contiguous 16B LDS reads.
static __device__ __forceinline__ v16bf load_b_frag(const __bf16* lds, int nbase, int lane) {
  const int half = lane >> 4, l16 = lane & 15;
  const __bf16* p = lds + (nbase + l16) * LDT + 16 * half;
  v16bf b;
  *(uint4*)&b       = *(const uint4*)p;
  *((uint4*)&b + 1) = *(const uint4*)(p + 8);
  return b;
}

// ---------------------------------------------------------------------------
// Prep kernels
// ---------------------------------------------------------------------------
__global__ void cvt_bf16_k(const float* __restrict__ in, __bf16* __restrict__ out, int n4) {
  int i = blockIdx.x * 256 + threadIdx.x;
  if (i >= n4) return;
  float4 f = ((const float4*)in)[i];
  BfPack4 h, l;
  split4(f, h, l);
  ((uint2*)out)[i] = h.u;
}

__global__ void split_bf16_k(const float* __restrict__ in, __bf16* __restrict__ hi,
                             __bf16* __restrict__ lo, int n4) {
  int i = blockIdx.x * 256 + threadIdx.x;
  if (i >= n4) return;
  float4 f = ((const float4*)in)[i];
  BfPack4 h, l;
  split4(f, h, l);
  ((uint2*)hi)[i] = h.u;
  ((uint2*)lo)[i] = l.u;
}

__global__ void copy_f32_k(const float* __restrict__ in, float* __restrict__ out, int n) {
  int i = blockIdx.x * 256 + threadIdx.x;
  if (i < n) out[i] = in[i];
}

// ---------------------------------------------------------------------------
// K1: x_gates[4096,1536] = embed[4096,512] @ w_ih^T + b_ih  (split-bf16, 3 WMMA)
// ---------------------------------------------------------------------------
__global__ __launch_bounds__(256) void gates_gemm(
    const float* __restrict__ X, const float* __restrict__ Wih,
    const float* __restrict__ bih, float* __restrict__ XG) {
  __shared__ __bf16 sAhi[128 * LDT], sAlo[128 * LDT], sBhi[128 * LDT], sBlo[128 * LDT];
  const int mt = blockIdx.x * 128, nt = blockIdx.y * 128;
  const int tid = threadIdx.x, wave = tid >> 5, lane = tid & 31;
  const int half = lane >> 4, l16 = lane & 15;
  v8f acc[8] = {};

  for (int k0 = 0; k0 < E_; k0 += 32) {
    // Stage 128x32 fp32 tiles of A and B, splitting into bf16 hi+lo.
#pragma unroll
    for (int it = 0; it < 4; ++it) {
      int idx = tid + it * 256, r = idx >> 3, c4 = idx & 7;
      float4 va = *(const float4*)(X   + (size_t)(mt + r) * E_ + k0 + c4 * 4);
      float4 vb = *(const float4*)(Wih + (size_t)(nt + r) * E_ + k0 + c4 * 4);
      BfPack4 ah, al, bh, bl;
      split4(va, ah, al);
      split4(vb, bh, bl);
      int o = r * LDT + c4 * 4;
      *(uint2*)&sAhi[o] = ah.u;
      *(uint2*)&sAlo[o] = al.u;
      *(uint2*)&sBhi[o] = bh.u;
      *(uint2*)&sBlo[o] = bl.u;
    }
    __syncthreads();
    v16bf ahi = load_a_frag(sAhi, wave * 16, lane);
    v16bf alo = load_a_frag(sAlo, wave * 16, lane);
#pragma unroll
    for (int j = 0; j < 8; ++j) {
      v16bf bhi = load_b_frag(sBhi, j * 16, lane);
      v16bf blo = load_b_frag(sBlo, j * 16, lane);
      acc[j] = wmma_bf16(ahi, bhi, acc[j]);
      acc[j] = wmma_bf16(ahi, blo, acc[j]);
      acc[j] = wmma_bf16(alo, bhi, acc[j]);
    }
    __syncthreads();
  }
  // C/D layout: VGPR i holds M = i + 8*half, N = l16 (+16j).
#pragma unroll
  for (int j = 0; j < 8; ++j) {
    int n = nt + j * 16 + l16;
    float bias = bih[n];
#pragma unroll
    for (int i = 0; i < 8; ++i) {
      int m = mt + wave * 16 + i + 8 * half;
      XG[(size_t)m * G3_ + n] = acc[j][i] + bias;
    }
  }
}

// ---------------------------------------------------------------------------
// K2: one GRU step. Grid (B/16, H/16), 32 threads (1 wave). Each block computes
// a 16x16 (b,h) tile of all three gate pre-activations h @ w_hh^T, then the
// full nonlinear update locally. h ping-pongs between buffers across launches.
// ---------------------------------------------------------------------------
__global__ __launch_bounds__(32) void gru_step(
    const float* __restrict__ XG, const __bf16* __restrict__ WhhHi,
    const __bf16* __restrict__ WhhLo, const float* __restrict__ bhh,
    const float* __restrict__ hin, float* __restrict__ hout,
    __bf16* __restrict__ hs, int s) {
  const int lane = threadIdx.x & 31, half = lane >> 4, l16 = lane & 15;
  const int bt = blockIdx.x * 16, ht = blockIdx.y * 16;
  v8f acc[3] = {};  // r, z, n gate pre-activation tiles

  for (int k0 = 0; k0 < H_; k0 += 32) {
    // A fragment straight from global fp32 h (L2-resident, tiny), split hi/lo.
    const float* ph = hin + (bt + l16) * H_ + k0 + 16 * half;
    v16bf ahi = {}, alo = {};
#pragma unroll
    for (int q = 0; q < 2; ++q)
#pragma unroll
      for (int t = 0; t < 8; ++t) {
        float f = ph[q * 8 + t];
        __bf16 hv = (__bf16)f;
        ahi[q * 8 + t] = hv;
        alo[q * 8 + t] = (__bf16)(f - (float)hv);
      }
#pragma unroll
    for (int g = 0; g < 3; ++g) {
      const __bf16* pw = WhhHi + (size_t)(g * H_ + ht + l16) * H_ + k0 + 16 * half;
      const __bf16* pl = WhhLo + (size_t)(g * H_ + ht + l16) * H_ + k0 + 16 * half;
      v16bf bhi, blo;
      *(uint4*)&bhi       = *(const uint4*)pw;
      *((uint4*)&bhi + 1) = *(const uint4*)(pw + 8);
      *(uint4*)&blo       = *(const uint4*)pl;
      *((uint4*)&blo + 1) = *(const uint4*)(pl + 8);
      acc[g] = wmma_bf16(ahi, bhi, acc[g]);
      acc[g] = wmma_bf16(ahi, blo, acc[g]);
      acc[g] = wmma_bf16(alo, bhi, acc[g]);
    }
  }

  const int n = ht + l16;
  const float br = bhh[n], bz = bhh[H_ + n], bn = bhh[2 * H_ + n];
#pragma unroll
  for (int i = 0; i < 8; ++i) {
    int b = bt + i + 8 * half;
    size_t row = (size_t)b * S_ + s;
    float xr = XG[row * G3_ + n];
    float xz = XG[row * G3_ + H_ + n];
    float xn = XG[row * G3_ + 2 * H_ + n];
    float hprev = hin[b * H_ + n];
    float r  = 1.0f / (1.0f + __expf(-(xr + acc[0][i] + br)));
    float z  = 1.0f / (1.0f + __expf(-(xz + acc[1][i] + bz)));
    float nn = tanhf(xn + r * (acc[2][i] + bn));
    float hnew = (1.0f - z) * nn + z * hprev;
    hout[b * H_ + n] = hnew;
    hs[row * H_ + n] = (__bf16)hnew;  // row = b*S+s -> logits row-major
  }
}

// ---------------------------------------------------------------------------
// K3: logits[4096,32000] = hs_bf16 @ w_out_bf16^T. 128x128 tile / workgroup,
// double-buffered LDS: stage chunk k+1 via async global->LDS while WMMAs
// consume chunk k; one s_wait_asynccnt + one barrier per chunk.
// fp32 accumulate, nontemporal stores for the 524 MB write-once output.
// ---------------------------------------------------------------------------
__global__ __launch_bounds__(256) void proj_gemm(
    const __bf16* __restrict__ A, const __bf16* __restrict__ Bw,
    float* __restrict__ out) {
  __shared__ __bf16 sA[2][128 * LDT], sB[2][128 * LDT];
  const int nt = blockIdx.x * 128, mt = blockIdx.y * 128;
  const int tid = threadIdx.x, wave = tid >> 5, lane = tid & 31;
  const int half = lane >> 4, l16 = lane & 15;
  v8f acc[8] = {};

  // Each thread stages 2x16B for A and B per chunk (512 uint4 per 128x32 tile).
  auto stage = [&](int kc, int buf) {
    const int k0 = kc * 32;
#pragma unroll
    for (int it = 0; it < 2; ++it) {
      int idx = tid + it * 256, r = idx >> 2, c4 = idx & 3;
      copy16_g2l(&sA[buf][r * LDT + c4 * 8], A  + (size_t)(mt + r) * H_ + k0 + c4 * 8);
      copy16_g2l(&sB[buf][r * LDT + c4 * 8], Bw + (size_t)(nt + r) * H_ + k0 + c4 * 8);
    }
  };

  const int NK = H_ / 32;  // 16 K-chunks
  stage(0, 0);
  for (int kc = 0; kc < NK; ++kc) {
    const int cur = kc & 1;
    wait_async_all();   // this wave's staged loads for buf[cur] have landed
    __syncthreads();    // all waves' loads landed; all waves done reading buf[cur^1]
    if (kc + 1 < NK) stage(kc + 1, cur ^ 1);  // overlap with compute below
    v16bf a = load_a_frag(sA[cur], wave * 16, lane);
#pragma unroll
    for (int j = 0; j < 8; ++j) {
      v16bf b = load_b_frag(sB[cur], j * 16, lane);
      acc[j] = wmma_bf16(a, b, acc[j]);
    }
  }
#pragma unroll
  for (int j = 0; j < 8; ++j)
#pragma unroll
    for (int i = 0; i < 8; ++i) {
      size_t m = (size_t)(mt + wave * 16 + i + 8 * half);
      size_t n = (size_t)(nt + j * 16 + l16);
      __builtin_nontemporal_store(acc[j][i], out + m * (size_t)V_ + n);
    }
}

// ---------------------------------------------------------------------------
extern "C" void kernel_launch(void* const* d_in, const int* in_sizes, int n_in,
                              void* d_out, int out_size, void* d_ws, size_t ws_size,
                              hipStream_t stream) {
  (void)in_sizes; (void)n_in; (void)out_size; (void)ws_size;
  const float* embed  = (const float*)d_in[0];  // [B,S,E]
  const float* hidden = (const float*)d_in[1];  // [1,B,H]
  const float* w_ih   = (const float*)d_in[2];  // [3H,E]
  const float* w_hh   = (const float*)d_in[3];  // [3H,H]
  const float* b_ih   = (const float*)d_in[4];  // [3H]
  const float* b_hh   = (const float*)d_in[5];  // [3H]
  const float* w_out  = (const float*)d_in[6];  // [V,H]
  float* logits = (float*)d_out;                // [B,S,V] row-major

  // Workspace carve-out (~65.5 MB total)
  char* base = (char*)d_ws;
  size_t off = 0;
  auto take = [&](size_t bytes) {
    char* p = base + off;
    off = (off + bytes + 255) & ~(size_t)255;
    return p;
  };
  float*  xg    = (float*) take(sizeof(float)  * (size_t)B_ * S_ * G3_);  // 25.2 MB
  float*  hbuf  = (float*) take(sizeof(float)  * 2 * B_ * H_);            // ping-pong h
  __bf16* hsb   = (__bf16*)take(sizeof(__bf16) * (size_t)B_ * S_ * H_);   // 4.2 MB
  __bf16* woutb = (__bf16*)take(sizeof(__bf16) * (size_t)V_ * H_);        // 32.8 MB
  __bf16* whhhi = (__bf16*)take(sizeof(__bf16) * (size_t)G3_ * H_);
  __bf16* whhlo = (__bf16*)take(sizeof(__bf16) * (size_t)G3_ * H_);

  // Prep: one-time conversions / init
  cvt_bf16_k  <<<(V_ * H_ / 4 + 255) / 256, 256, 0, stream>>>(w_out, woutb, V_ * H_ / 4);
  split_bf16_k<<<(G3_ * H_ / 4 + 255) / 256, 256, 0, stream>>>(w_hh, whhhi, whhlo, G3_ * H_ / 4);
  copy_f32_k  <<<(B_ * H_ + 255) / 256, 256, 0, stream>>>(hidden, hbuf, B_ * H_);

  // K1: input gate GEMM
  gates_gemm<<<dim3((B_ * S_) / 128, G3_ / 128), 256, 0, stream>>>(embed, w_ih, b_ih, xg);

  // K2: sequential GRU scan, one launch per step (stream/graph ordered)
  for (int s = 0; s < S_; ++s) {
    const float* hin = hbuf + (s & 1) * B_ * H_;
    float* hout      = hbuf + ((s + 1) & 1) * B_ * H_;
    gru_step<<<dim3(B_ / 16, H_ / 16), 32, 0, stream>>>(xg, whhhi, whhlo, b_hh, hin, hout, hsb, s);
  }

  // K3: vocab projection (dominant: 134 GFLOP, 524 MB out)
  proj_gemm<<<dim3(V_ / 128, (B_ * S_) / 128), 256, 0, stream>>>(hsb, woutb, logits);
}
